// HMMStage_46042049413823
// MI455X (gfx1250) — matrix operand order repaired and checked by reading, not compiled, not run on hardware
//
#include <hip/hip_runtime.h>
#include <hip/hip_fp16.h>

typedef _Float16 h16_t;
typedef __attribute__((ext_vector_type(16))) _Float16 v16h;
typedef __attribute__((ext_vector_type(8)))  float    v8f;

#define BL   32768   // 16*2048
#define LSEQ 2048
#define DM   64
#define DFF  2048

__device__ __forceinline__ v8f wmma16(v16h a, v16h b, v8f c) {
  return __builtin_amdgcn_wmma_f32_16x16x32_f16(false, a, false, b, (short)0, c, false, false);
}

// A-layout 16x32 f16 tile from row-major half buffer (ld halves per row).
// lane m=lane&15 holds row m; elems 0..15 -> K=16*hi+e
__device__ __forceinline__ v16h loadA_h16(const h16_t* base, int ld, int kOff) {
  int lane = threadIdx.x & 31;
  int m = lane & 15, hi = lane >> 4;
  const h16_t* p0 = base + (size_t)m * ld + kOff + hi * 16;
  const h16_t* p1 = p0 + 8;
  v16h a;
#pragma unroll
  for (int e = 0; e < 8; ++e) a[e] = p0[e];
#pragma unroll
  for (int e = 0; e < 8; ++e) a[8 + e] = p1[e];
  return a;
}

// B-layout 32x16: B[k][n] = W[nOff+n][kOff + 16*hi + e], W row-major f32, leading dim ld
__device__ __forceinline__ v16h loadB_w32(const float* W, int ld, int nOff, int kOff) {
  int lane = threadIdx.x & 31;
  int n = lane & 15, hi = lane >> 4;
  const float* p = W + (size_t)(nOff + n) * ld + kOff + hi * 16;
  v16h b;
#pragma unroll
  for (int e = 0; e < 16; ++e) b[e] = (h16_t)p[e];
  return b;
}

// same but from a row-major f16 buffer (e.g. LDS); source stores W[n][k]
__device__ __forceinline__ v16h loadB_h16(const h16_t* W, int ld, int nOff, int kOff) {
  int lane = threadIdx.x & 31;
  int n = lane & 15, hi = lane >> 4;
  const h16_t* p = W + (size_t)(nOff + n) * ld + kOff + hi * 16;
  v16h b;
#pragma unroll
  for (int e = 0; e < 16; ++e) b[e] = p[e];
  return b;
}

// B-layout from natural [k][n] f16 buffer (strided gather): B[k][n] = V[k][nOff+n]
__device__ __forceinline__ v16h loadB_h16_nat(const h16_t* V, int ld, int nOff) {
  int lane = threadIdx.x & 31;
  int n = lane & 15, hi = lane >> 4;
  const h16_t* p = V + (size_t)(16 * hi) * ld + nOff + n;
  v16h b;
#pragma unroll
  for (int e = 0; e < 16; ++e) b[e] = p[(size_t)e * ld];
  return b;
}

// CDNA5 async copy: 16 bytes/lane global -> LDS, tracked by ASYNCcnt
__device__ __forceinline__ void async_b128(unsigned ldsOff, const void* gaddr) {
  asm volatile("global_load_async_to_lds_b128 %0, %1, off"
               :: "v"(ldsOff), "v"(gaddr) : "memory");
}

__device__ __forceinline__ float redmax16(float v) {
#pragma unroll
  for (int m = 1; m < 16; m <<= 1) v = fmaxf(v, __shfl_xor(v, m, 32));
  return v;
}
__device__ __forceinline__ float redsum16(float v) {
#pragma unroll
  for (int m = 1; m < 16; m <<= 1) v += __shfl_xor(v, m, 32);
  return v;
}
__device__ __forceinline__ float lse3(float a, float b, float c) {
  float m = fmaxf(a, fmaxf(b, c));
  return m + logf(expf(a - m) + expf(b - m) + expf(c - m));
}

// ---------------- 1. embed: h = x @ embed_w^T + embed_b -----------------
__global__ void embed_kernel(const float* __restrict__ x, const float* __restrict__ ew,
                             const float* __restrict__ eb, float* __restrict__ h32,
                             h16_t* __restrict__ h16) {
  int idx = blockIdx.x * 256 + threadIdx.x;   // row*64 + dd
  int row = idx >> 6, dd = idx & 63;
  const float* xr = x + (size_t)row * 3;
  float v = eb[dd] + xr[0] * ew[dd * 3 + 0] + xr[1] * ew[dd * 3 + 1] + xr[2] * ew[dd * 3 + 2];
  h32[idx] = v;
  h16[idx] = (h16_t)v;
}

// ---------------- 2. q,k,v projections (WMMA) -----------------
__global__ void __launch_bounds__(256) qkv_kernel(
    const h16_t* __restrict__ h16,
    const float* __restrict__ wq, const float* __restrict__ bq,
    const float* __restrict__ wk, const float* __restrict__ bk,
    const float* __restrict__ wv, const float* __restrict__ bv,
    h16_t* __restrict__ q16, h16_t* __restrict__ k16, h16_t* __restrict__ v16o) {
  int wave = threadIdx.x >> 5;
  int rowBase = (blockIdx.x * 8 + wave) * 16;
  int lane = threadIdx.x & 31;
  int n = lane & 15, hi = lane >> 4;
  v16h a0 = loadA_h16(h16 + (size_t)rowBase * DM, DM, 0);
  v16h a1 = loadA_h16(h16 + (size_t)rowBase * DM, DM, 32);
  const float* Ws[3] = {wq, wk, wv};
  const float* Bs[3] = {bq, bk, bv};
  h16_t* Os[3] = {q16, k16, v16o};
#pragma unroll
  for (int mtx = 0; mtx < 3; ++mtx) {
    float scl = (mtx == 0) ? 0.125f : 1.0f;   // fold 1/sqrt(64) into q
#pragma unroll
    for (int nt = 0; nt < 4; ++nt) {
      v8f c = {};
      c = wmma16(a0, loadB_w32(Ws[mtx], DM, nt * 16, 0), c);
      c = wmma16(a1, loadB_w32(Ws[mtx], DM, nt * 16, 32), c);
      int col = nt * 16 + n;
      float bias = Bs[mtx][col];
#pragma unroll
      for (int r = 0; r < 8; ++r) {
        int row = rowBase + r + 8 * hi;
        Os[mtx][(size_t)row * DM + col] = (h16_t)((c[r] + bias) * scl);
      }
    }
  }
}

// ------ 3. flash attention: async double-buffered K/V, WMMA scores + WMMA P*V ------
__global__ void __launch_bounds__(256) attn_kernel(
    const h16_t* __restrict__ q16, const h16_t* __restrict__ k16,
    const h16_t* __restrict__ v16i, h16_t* __restrict__ o16) {
  __shared__ __align__(16) h16_t Klds[2][32 * DM];  // [buf][key][d]
  __shared__ __align__(16) h16_t Vlds[2][32 * DM];  // [buf][key][d]
  __shared__ __align__(16) h16_t Pst[8][16 * 32];   // per-wave P staging
  int tid = threadIdx.x;
  int wave = tid >> 5;
  int lane = tid & 31;
  int n = lane & 15, hi = lane >> 4;
  int batch = blockIdx.x >> 4;
  int qBase = batch * LSEQ + (blockIdx.x & 15) * 128 + wave * 16;
  v16h aq0 = loadA_h16(q16 + (size_t)qBase * DM, DM, 0);
  v16h aq1 = loadA_h16(q16 + (size_t)qBase * DM, DM, 32);
  v8f o0 = {}, o1 = {}, o2 = {}, o3 = {};
  float mrun[8], lrun[8];
#pragma unroll
  for (int r = 0; r < 8; ++r) { mrun[r] = -1e30f; lrun[r] = 0.f; }

  const char* kbase = (const char*)(k16 + (size_t)batch * LSEQ * DM);
  const char* vbase = (const char*)(v16i + (size_t)batch * LSEQ * DM);
  unsigned kldsOff0 = (unsigned)(size_t)&Klds[0][0] + tid * 16;
  unsigned vldsOff0 = (unsigned)(size_t)&Vlds[0][0] + tid * 16;
  const unsigned bufStride = 32 * DM * 2;        // bytes per buffer
  // prologue: issue block 0 into buffer 0 (each thread copies 16B of K and V)
  async_b128(kldsOff0, kbase + tid * 16);
  async_b128(vldsOff0, vbase + tid * 16);

#pragma unroll 1
  for (int kb = 0; kb < LSEQ / 32; ++kb) {
    int buf = kb & 1;
    asm volatile("s_wait_asynccnt 0" ::: "memory");  // my inflight copies done
    __syncthreads();                                 // everyone done with buf^1
    if (kb + 1 < LSEQ / 32) {                        // prefetch next block async
      size_t goff = (size_t)(kb + 1) * 32 * DM * 2 + tid * 16;
      unsigned lo = (buf ^ 1) ? bufStride : 0u;
      async_b128(kldsOff0 + lo, kbase + goff);
      async_b128(vldsOff0 + lo, vbase + goff);
    }
    const h16_t* Kb = &Klds[buf][0];
    const h16_t* Vb = &Vlds[buf][0];
    // S[16q x 32keys] = q . k^T  (q pre-scaled)
    v8f s0 = {}, s1 = {};
    s0 = wmma16(aq0, loadB_h16(Kb, DM, 0, 0), s0);
    s0 = wmma16(aq1, loadB_h16(Kb, DM, 0, 32), s0);
    s1 = wmma16(aq0, loadB_h16(Kb, DM, 16, 0), s1);
    s1 = wmma16(aq1, loadB_h16(Kb, DM, 16, 32), s1);
    float p0[8], p1[8], scO[8];
#pragma unroll
    for (int r = 0; r < 8; ++r) {
      float mx = redmax16(fmaxf(s0[r], s1[r]));
      float nm = fmaxf(mrun[r], mx);
      float sc = __expf(mrun[r] - nm);
      p0[r] = __expf(s0[r] - nm);
      p1[r] = __expf(s1[r] - nm);
      float rs = redsum16(p0[r] + p1[r]);
      lrun[r] = lrun[r] * sc + rs;
      mrun[r] = nm;
      scO[r] = sc;
    }
#pragma unroll
    for (int r = 0; r < 8; ++r) { o0[r] *= scO[r]; o1[r] *= scO[r]; o2[r] *= scO[r]; o3[r] *= scO[r]; }
    // stage P (C-layout -> row-major LDS), then reload in A-layout
#pragma unroll
    for (int r = 0; r < 8; ++r) {
      int row = r + 8 * hi;
      Pst[wave][row * 32 + n] = (h16_t)p0[r];
      Pst[wave][row * 32 + 16 + n] = (h16_t)p1[r];
    }
    asm volatile("s_wait_dscnt 0" ::: "memory");   // wave-local LDS RAW fence
    v16h aP = loadA_h16(&Pst[wave][0], 32, 0);
    o0 = wmma16(aP, loadB_h16_nat(Vb, DM, 0), o0);
    o1 = wmma16(aP, loadB_h16_nat(Vb, DM, 16), o1);
    o2 = wmma16(aP, loadB_h16_nat(Vb, DM, 32), o2);
    o3 = wmma16(aP, loadB_h16_nat(Vb, DM, 48), o3);
  }
#pragma unroll
  for (int r = 0; r < 8; ++r) {
    float inv = 1.0f / lrun[r];
    size_t row = (size_t)(qBase + r + 8 * hi) * DM;
    o16[row + n] = (h16_t)(o0[r] * inv);
    o16[row + 16 + n] = (h16_t)(o1[r] * inv);
    o16[row + 32 + n] = (h16_t)(o2[r] * inv);
    o16[row + 48 + n] = (h16_t)(o3[r] * inv);
  }
}

// ---------------- 4. output proj + residual + LN1 -----------------
__global__ void __launch_bounds__(256) oproj_ln1_kernel(
    const h16_t* __restrict__ ao16, const float* __restrict__ wo, const float* __restrict__ bo,
    const float* __restrict__ h32, const float* __restrict__ g, const float* __restrict__ bln,
    float* __restrict__ h1_32, h16_t* __restrict__ h1_16) {
  int wave = threadIdx.x >> 5;
  int rowBase = (blockIdx.x * 8 + wave) * 16;
  int lane = threadIdx.x & 31;
  int n = lane & 15, hi = lane >> 4;
  v16h a0 = loadA_h16(ao16 + (size_t)rowBase * DM, DM, 0);
  v16h a1 = loadA_h16(ao16 + (size_t)rowBase * DM, DM, 32);
  v8f c[4];
#pragma unroll
  for (int nt = 0; nt < 4; ++nt) {
    v8f t = {};
    t = wmma16(a0, loadB_w32(wo, DM, nt * 16, 0), t);
    t = wmma16(a1, loadB_w32(wo, DM, nt * 16, 32), t);
    int col = nt * 16 + n;
    float bias = bo[col];
#pragma unroll
    for (int r = 0; r < 8; ++r)
      t[r] = t[r] + bias + h32[(size_t)(rowBase + r + 8 * hi) * DM + col];
    c[nt] = t;
  }
#pragma unroll
  for (int r = 0; r < 8; ++r) {
    float s = c[0][r] + c[1][r] + c[2][r] + c[3][r];
    float ss = c[0][r] * c[0][r] + c[1][r] * c[1][r] + c[2][r] * c[2][r] + c[3][r] * c[3][r];
    float mean = redsum16(s) * (1.f / 64.f);
    float var = redsum16(ss) * (1.f / 64.f) - mean * mean;
    float rstd = rsqrtf(var + 1e-5f);
    size_t row = (size_t)(rowBase + r + 8 * hi) * DM;
#pragma unroll
    for (int nt = 0; nt < 4; ++nt) {
      int col = nt * 16 + n;
      float v = (c[nt][r] - mean) * rstd * g[col] + bln[col];
      h1_32[row + col] = v;
      h1_16[row + col] = (h16_t)v;
    }
  }
}

// ---------------- 5. fused FFN (relu(h@W1^T+b1)@W2^T+b2) + residual + LN2 + relu ----
__global__ void __launch_bounds__(256) ffn_kernel(
    const h16_t* __restrict__ h1_16,
    const float* __restrict__ w1, const float* __restrict__ b1,
    const float* __restrict__ w2, const float* __restrict__ b2,
    const float* __restrict__ h1_32, const float* __restrict__ g, const float* __restrict__ bln,
    float* __restrict__ h2_32) {
  __shared__ __align__(16) h16_t Ust[8][16 * 32];
  int wave = threadIdx.x >> 5;
  int rowBase = (blockIdx.x * 8 + wave) * 16;
  int lane = threadIdx.x & 31;
  int n = lane & 15, hi = lane >> 4;
  v16h a0 = loadA_h16(h1_16 + (size_t)rowBase * DM, DM, 0);
  v16h a1 = loadA_h16(h1_16 + (size_t)rowBase * DM, DM, 32);
  v8f c0 = {}, c1 = {}, c2 = {}, c3 = {};
#pragma unroll 1
  for (int fc = 0; fc < DFF / 32; ++fc) {
    int f0 = fc * 32;
    if (fc + 1 < DFF / 32) __builtin_prefetch(w1 + (size_t)(f0 + 32) * DM, 0, 3);
    v8f u0 = {}, u1 = {};
    u0 = wmma16(a0, loadB_w32(w1, DM, f0, 0), u0);
    u0 = wmma16(a1, loadB_w32(w1, DM, f0, 32), u0);
    u1 = wmma16(a0, loadB_w32(w1, DM, f0 + 16, 0), u1);
    u1 = wmma16(a1, loadB_w32(w1, DM, f0 + 16, 32), u1);
    float bias0 = b1[f0 + n], bias1 = b1[f0 + 16 + n];
#pragma unroll
    for (int r = 0; r < 8; ++r) {
      int row = r + 8 * hi;
      Ust[wave][row * 32 + n] = (h16_t)fmaxf(u0[r] + bias0, 0.f);
      Ust[wave][row * 32 + 16 + n] = (h16_t)fmaxf(u1[r] + bias1, 0.f);
    }
    asm volatile("s_wait_dscnt 0" ::: "memory");
    v16h aU = loadA_h16(&Ust[wave][0], 32, 0);
    c0 = wmma16(aU, loadB_w32(w2, DFF, 0, f0), c0);
    c1 = wmma16(aU, loadB_w32(w2, DFF, 16, f0), c1);
    c2 = wmma16(aU, loadB_w32(w2, DFF, 32, f0), c2);
    c3 = wmma16(aU, loadB_w32(w2, DFF, 48, f0), c3);
  }
  v8f cc[4] = {c0, c1, c2, c3};
#pragma unroll
  for (int nt = 0; nt < 4; ++nt) {
    int col = nt * 16 + n;
    float bias = b2[col];
#pragma unroll
    for (int r = 0; r < 8; ++r)
      cc[nt][r] += bias + h1_32[(size_t)(rowBase + r + 8 * hi) * DM + col];
  }
#pragma unroll
  for (int r = 0; r < 8; ++r) {
    float s = cc[0][r] + cc[1][r] + cc[2][r] + cc[3][r];
    float ss = cc[0][r] * cc[0][r] + cc[1][r] * cc[1][r] + cc[2][r] * cc[2][r] + cc[3][r] * cc[3][r];
    float mean = redsum16(s) * (1.f / 64.f);
    float var = redsum16(ss) * (1.f / 64.f) - mean * mean;
    float rstd = rsqrtf(var + 1e-5f);
    size_t row = (size_t)(rowBase + r + 8 * hi) * DM;
#pragma unroll
    for (int nt = 0; nt < 4; ++nt) {
      int col = nt * 16 + n;
      float v = (cc[nt][r] - mean) * rstd * g[col] + bln[col];
      h2_32[row + col] = fmaxf(v, 0.f);
    }
  }
}

// ---------------- 6. emission logits -----------------
__global__ void emis_kernel(const float* __restrict__ h2, const float* __restrict__ hw,
                            const float* __restrict__ hb, float* __restrict__ emis) {
  int row = blockIdx.x * 256 + threadIdx.x;
  if (row >= BL) return;
  const float* hr = h2 + (size_t)row * DM;
  float e0 = hb[0], e1 = hb[1], e2 = hb[2];
#pragma unroll 8
  for (int d = 0; d < DM; ++d) {
    float v = hr[d];
    e0 += v * hw[d];
    e1 += v * hw[DM + d];
    e2 += v * hw[2 * DM + d];
  }
  emis[row * 3 + 0] = e0;
  emis[row * 3 + 1] = e1;
  emis[row * 3 + 2] = e2;
}

// ---------------- 7. mean over L -----------------
__global__ void mean_kernel(const float* __restrict__ h2, float* __restrict__ out) {
  int b = blockIdx.x;
  int dd = threadIdx.x;   // 64 threads
  float s = 0.f;
  for (int l = 0; l < LSEQ; ++l) s += h2[((size_t)b * LSEQ + l) * DM + dd];
  out[16 + b * DM + dd] = s * (1.f / (float)LSEQ);
}

// ---------------- 8. HMM forward (log domain), sequential scan -----------------
__global__ void hmm_kernel(const float* __restrict__ emis, const float* __restrict__ sl,
                           const float* __restrict__ tl, float* __restrict__ out) {
  int b = threadIdx.x;
  if (b >= 16) return;
  float z = lse3(sl[0], sl[1], sl[2]);
  float ls0 = sl[0] - z, ls1 = sl[1] - z, ls2 = sl[2] - z;
  float lt[3][3];
#pragma unroll
  for (int i = 0; i < 3; ++i) {
    float zi = lse3(tl[i * 3 + 0], tl[i * 3 + 1], tl[i * 3 + 2]);
#pragma unroll
    for (int j = 0; j < 3; ++j) lt[i][j] = tl[i * 3 + j] - zi;
  }
  const float* eb = emis + (size_t)b * LSEQ * 3;
  float la0 = ls0 + eb[0], la1 = ls1 + eb[1], la2 = ls2 + eb[2];
  for (int l = 1; l < LSEQ; ++l) {
    const float* e = eb + l * 3;
    float n0 = lse3(la0 + lt[0][0], la1 + lt[1][0], la2 + lt[2][0]) + e[0];
    float n1 = lse3(la0 + lt[0][1], la1 + lt[1][1], la2 + lt[2][1]) + e[1];
    float n2 = lse3(la0 + lt[0][2], la1 + lt[1][2], la2 + lt[2][2]) + e[2];
    la0 = n0; la1 = n1; la2 = n2;
  }
  float lp = lse3(la0, la1, la2);
  out[b] = 1.0f / (1.0f + expf(-lp));
}

extern "C" void kernel_launch(void* const* d_in, const int* in_sizes, int n_in,
                              void* d_out, int out_size, void* d_ws, size_t ws_size,
                              hipStream_t stream) {
  (void)in_sizes; (void)n_in; (void)out_size; (void)ws_size;
  const float* x       = (const float*)d_in[0];
  const float* embed_w = (const float*)d_in[1];
  const float* embed_b = (const float*)d_in[2];
  const float* wq = (const float*)d_in[3];  const float* bq = (const float*)d_in[4];
  const float* wk = (const float*)d_in[5];  const float* bk = (const float*)d_in[6];
  const float* wv = (const float*)d_in[7];  const float* bv = (const float*)d_in[8];
  const float* wo = (const float*)d_in[9];  const float* bo = (const float*)d_in[10];
  const float* ln1_g = (const float*)d_in[11]; const float* ln1_b = (const float*)d_in[12];
  const float* ff1_w = (const float*)d_in[13]; const float* ff1_b = (const float*)d_in[14];
  const float* ff2_w = (const float*)d_in[15]; const float* ff2_b = (const float*)d_in[16];
  const float* ln2_g = (const float*)d_in[17]; const float* ln2_b = (const float*)d_in[18];
  const float* head_w = (const float*)d_in[19]; const float* head_b = (const float*)d_in[20];
  const float* start_logits = (const float*)d_in[21];
  const float* trans_logits = (const float*)d_in[22];

  char* ws = (char*)d_ws;
  const size_t MB = 1ull << 20;
  float*  h32   = (float*)(ws + 0 * MB);    // 8 MB
  float*  h1_32 = (float*)(ws + 8 * MB);    // 8 MB
  float*  h2_32 = (float*)(ws + 16 * MB);   // 8 MB
  h16_t*  h16   = (h16_t*)(ws + 24 * MB);   // 4 MB
  h16_t*  q16   = (h16_t*)(ws + 28 * MB);   // 4 MB
  h16_t*  k16   = (h16_t*)(ws + 32 * MB);   // 4 MB
  h16_t*  v16b  = (h16_t*)(ws + 36 * MB);   // 4 MB
  h16_t*  ao16  = (h16_t*)(ws + 40 * MB);   // 4 MB
  h16_t*  h1_16 = (h16_t*)(ws + 44 * MB);   // 4 MB
  float*  emis  = (float*)(ws + 48 * MB);   // 384 KB
  float*  out   = (float*)d_out;

  embed_kernel<<<(BL * DM) / 256, 256, 0, stream>>>(x, embed_w, embed_b, h32, h16);
  qkv_kernel<<<BL / 128, 256, 0, stream>>>(h16, wq, bq, wk, bk, wv, bv, q16, k16, v16b);
  attn_kernel<<<16 * (LSEQ / 128), 256, 0, stream>>>(q16, k16, v16b, ao16);
  oproj_ln1_kernel<<<BL / 128, 256, 0, stream>>>(ao16, wo, bo, h32, ln1_g, ln1_b, h1_32, h1_16);
  ffn_kernel<<<BL / 128, 256, 0, stream>>>(h1_16, ff1_w, ff1_b, ff2_w, ff2_b,
                                           h1_32, ln2_g, ln2_b, h2_32);
  emis_kernel<<<BL / 256, 256, 0, stream>>>(h2_32, head_w, head_b, emis);
  mean_kernel<<<16, 64, 0, stream>>>(h2_32, out);
  hmm_kernel<<<1, 32, 0, stream>>>(emis, start_logits, trans_logits, out);
}
